// SutraV061_37477884625520
// MI455X (gfx1250) — matrix operand order repaired and checked by reading, not compiled, not run on hardware
//
#include <hip/hip_runtime.h>
#include <cstdint>
#include <cstddef>

#define NMODES 4
#define NSTAGES 7
#define CH 128
#define PDIM 32
#define MAXP 12

typedef __bf16 bf16_t;
typedef __attribute__((ext_vector_type(16))) __bf16 bf16x16;
typedef __attribute__((ext_vector_type(8)))  __bf16 bf16x8;
typedef __attribute__((ext_vector_type(8)))  float  f32x8;

// ---------------------------------------------------------------------------
// Weight f32 -> bf16, compacting row stride (kin -> kout) and zero-padding
// extra rows (rows_in -> rows_out) so GEMM B-loads never need bounds checks.
// ---------------------------------------------------------------------------
__global__ void k_cvt_bf16(const float* __restrict__ in, bf16_t* __restrict__ out,
                           int rows_in, int rows_out, int kin, int kout) {
  int idx = blockIdx.x * blockDim.x + threadIdx.x;
  int total = rows_out * kout;
  if (idx >= total) return;
  int r = idx / kout, k = idx - r * kout;
  out[idx] = (r < rows_in) ? (bf16_t)in[(size_t)r * kin + k] : (bf16_t)0.0f;
}

// ---------------------------------------------------------------------------
// One-block precompute: effective content+bilinear weight cw (4x128),
// constant logits (4), and the 4 "tail" columns of mlp_w1 (128x4).
// ---------------------------------------------------------------------------
__global__ void k_prep_small(const float* __restrict__ pass_embed,
                             const float* __restrict__ ch_w, const float* __restrict__ ch_b,
                             const float* __restrict__ ph_w, const float* __restrict__ ph_b,
                             const float* __restrict__ bil_w, const float* __restrict__ mode_bias,
                             const float* __restrict__ mlp_w1,
                             const int* __restrict__ pass_idx_p,
                             float* __restrict__ cw, float* __restrict__ constl,
                             float* __restrict__ tail) {
  int j = threadIdx.x;            // 128 threads
  int pi = *pass_idx_p;
  const float* u = pass_embed + (size_t)pi * PDIM;
  #pragma unroll
  for (int m = 0; m < NMODES; ++m) {
    float s = ch_w[m * CH + j];
    for (int k = 0; k < PDIM; ++k) s += u[k] * bil_w[(size_t)(m * PDIM + k) * CH + j];
    cw[m * CH + j] = s;
  }
  if (j < NMODES) {
    float s = ch_b[j] + ph_b[j] + mode_bias[j];
    for (int k = 0; k < PDIM; ++k) s += u[k] * ph_w[j * PDIM + k];
    constl[j] = s;
  }
  #pragma unroll
  for (int c = 0; c < 4; ++c) tail[j * 4 + c] = mlp_w1[(size_t)j * 1540 + 1536 + c];
}

// ---------------------------------------------------------------------------
// Per-token prep: LN stats + log-lambda stats + dmu; emit bf16 A matrices.
//   h_bf  : (BT,768)   bf16 copy of h
//   r_bf  : (BT,1536)  [ LN(h) | emb_x ] in bf16
//   stats : (BT,4)     { mean_ll, std_ll, dmu, 0 }
// ---------------------------------------------------------------------------
__device__ inline float block_reduce_sum(float v, float* red) {
  int tid = threadIdx.x;
  red[tid] = v; __syncthreads();
  for (int s = blockDim.x >> 1; s > 0; s >>= 1) {
    if (tid < s) red[tid] += red[tid + s];
    __syncthreads();
  }
  float r = red[0]; __syncthreads();
  return r;
}

__global__ void k_prep_token(const float* __restrict__ h, const float* __restrict__ emb,
                             const float* __restrict__ lam, const float* __restrict__ mu_prev,
                             const float* __restrict__ ln_g, const float* __restrict__ ln_b,
                             bf16_t* __restrict__ h_bf, bf16_t* __restrict__ r_bf,
                             float* __restrict__ stats) {
  const int D = 768;
  int t = blockIdx.x, tid = threadIdx.x;
  const float* hp = h       + (size_t)t * D;
  const float* ep = emb     + (size_t)t * D;
  const float* lp = lam     + (size_t)t * D;
  const float* mp = mu_prev + (size_t)t * D;

  float sh = 0.f, sh2 = 0.f, sll = 0.f, sll2 = 0.f, sd = 0.f;
  for (int k = tid; k < D; k += blockDim.x) {
    float x = hp[k];
    sh += x; sh2 += x * x;
    float ll = __logf(fmaxf(lp[k], 1e-6f));
    sll += ll; sll2 += ll * ll;
    float d = x - mp[k];
    sd += d * d;
  }
  __shared__ float red[256];
  sh   = block_reduce_sum(sh,   red);
  sh2  = block_reduce_sum(sh2,  red);
  sll  = block_reduce_sum(sll,  red);
  sll2 = block_reduce_sum(sll2, red);
  sd   = block_reduce_sum(sd,   red);

  const float invD = 1.0f / (float)D;
  float mean = sh * invD;
  float var  = fmaxf(sh2 * invD - mean * mean, 0.f);
  float rstd = rsqrtf(var + 1e-5f);
  float mll  = sll * invD;
  float vll  = fmaxf((sll2 - (float)D * mll * mll) / (float)(D - 1), 0.f);
  float stdll = fmaxf(sqrtf(vll), 1e-6f);
  float dmu  = sqrtf(sd * invD + 1e-8f);

  if (tid == 0) {
    stats[t * 4 + 0] = mll;
    stats[t * 4 + 1] = stdll;
    stats[t * 4 + 2] = dmu;
    stats[t * 4 + 3] = 0.f;
  }
  for (int k = tid; k < D; k += blockDim.x) {
    float x = hp[k];
    h_bf[(size_t)t * D + k] = (bf16_t)x;
    r_bf[(size_t)t * 1536 + k]       = (bf16_t)((x - mean) * rstd * ln_g[k] + ln_b[k]);
    r_bf[(size_t)t * 1536 + 768 + k] = (bf16_t)ep[k];
  }
}

// ---------------------------------------------------------------------------
// BF16 WMMA GEMM:  out(M,N) = A(M,K) * W(Npad,K)^T + bias
//   One wave per 16x64 output tile (4 n-subtiles), K stepped by 32.
//   Software-pipelined: fragments for step k+1 are loaded into distinct
//   registers while the 4 WMMAs for step k execute back-to-back, so only one
//   loadcnt wait is needed per iteration. Last step peeled (no OOB prefetch).
//   W is zero-padded to a multiple of 64 rows -> no bounds checks in the hot
//   loop. Store epilogue masks cols >= N.
//   MODE 0: silu -> bf16 out                        (GEMM1)
//   MODE 1: + rank-4 stats tail, silu -> bf16 out   (GEMM3)
//   MODE 2: f32 out                                 (GEMM2/GEMM4)
// ---------------------------------------------------------------------------
template <int MODE>
__global__ __launch_bounds__(32)
void k_gemm_wmma(const bf16_t* __restrict__ A, const bf16_t* __restrict__ W,
                 const float* __restrict__ bias, void* __restrict__ outp,
                 int M, int N, int K,
                 const float* __restrict__ stats, const float* __restrict__ tail,
                 const int* __restrict__ pass_idx_p) {
  int lane  = threadIdx.x;
  int row0  = blockIdx.x * 16;
  int n0b   = blockIdx.y * 64;
  int nlane = lane & 15;     // C column / A row / W row within tile
  int half  = lane >> 4;

  f32x8 acc[4];
  #pragma unroll
  for (int t = 0; t < 4; ++t)
    #pragma unroll
    for (int i = 0; i < 8; ++i) acc[t][i] = 0.f;

  const bf16_t* arow = A + (size_t)(row0 + nlane) * K;
  const bf16_t* wrow = W + (size_t)(n0b + nlane) * K + half * 16;
  const size_t wtile = (size_t)16 * K;

  auto load_a = [&](int k0) -> bf16x16 {
    bf16x8 lo = *(const bf16x8*)(arow + k0 + half * 16);
    bf16x8 hi = *(const bf16x8*)(arow + k0 + half * 16 + 8);
    bf16x16 a;
    #pragma unroll
    for (int i = 0; i < 8; ++i) { a[i] = lo[i]; a[8 + i] = hi[i]; }
    return a;
  };
  auto load_b = [&](int k0, int t) -> bf16x16 {
    const bf16_t* wp = wrow + (size_t)t * wtile + k0;
    bf16x8 b0 = *(const bf16x8*)(wp);
    bf16x8 b1 = *(const bf16x8*)(wp + 8);
    bf16x16 b;
    #pragma unroll
    for (int i = 0; i < 8; ++i) { b[i] = b0[i]; b[8 + i] = b1[i]; }
    return b;
  };

  // prologue: fragments for k0 = 0
  bf16x16 a_cur = load_a(0);
  bf16x16 b_cur0 = load_b(0, 0);
  bf16x16 b_cur1 = load_b(0, 1);
  bf16x16 b_cur2 = load_b(0, 2);
  bf16x16 b_cur3 = load_b(0, 3);

  for (int k0 = 0; k0 + 32 < K; k0 += 32) {
    // prefetch next step into fresh registers (independent of current WMMAs)
    bf16x16 a_nxt  = load_a(k0 + 32);
    bf16x16 b_nxt0 = load_b(k0 + 32, 0);
    bf16x16 b_nxt1 = load_b(k0 + 32, 1);
    bf16x16 b_nxt2 = load_b(k0 + 32, 2);
    bf16x16 b_nxt3 = load_b(k0 + 32, 3);

    acc[0] = __builtin_amdgcn_wmma_f32_16x16x32_bf16(false, a_cur, false, b_cur0, (short)0, acc[0], false, false);
    acc[1] = __builtin_amdgcn_wmma_f32_16x16x32_bf16(false, a_cur, false, b_cur1, (short)0, acc[1], false, false);
    acc[2] = __builtin_amdgcn_wmma_f32_16x16x32_bf16(false, a_cur, false, b_cur2, (short)0, acc[2], false, false);
    acc[3] = __builtin_amdgcn_wmma_f32_16x16x32_bf16(false, a_cur, false, b_cur3, (short)0, acc[3], false, false);

    a_cur = a_nxt;
    b_cur0 = b_nxt0; b_cur1 = b_nxt1; b_cur2 = b_nxt2; b_cur3 = b_nxt3;
  }
  // peeled final step
  acc[0] = __builtin_amdgcn_wmma_f32_16x16x32_bf16(false, a_cur, false, b_cur0, (short)0, acc[0], false, false);
  acc[1] = __builtin_amdgcn_wmma_f32_16x16x32_bf16(false, a_cur, false, b_cur1, (short)0, acc[1], false, false);
  acc[2] = __builtin_amdgcn_wmma_f32_16x16x32_bf16(false, a_cur, false, b_cur2, (short)0, acc[2], false, false);
  acc[3] = __builtin_amdgcn_wmma_f32_16x16x32_bf16(false, a_cur, false, b_cur3, (short)0, acc[3], false, false);

  // epilogue
  float s0[8], s1[8], s2[8];
  float pf = 0.f;
  if (MODE == 1) {
    pf = (float)(*pass_idx_p) * (1.0f / (float)(MAXP - 1));
    #pragma unroll
    for (int v = 0; v < 8; ++v) {
      int r = row0 + half * 8 + v;
      s0[v] = stats[r * 4 + 0];
      s1[v] = stats[r * 4 + 1];
      s2[v] = stats[r * 4 + 2];
    }
  }

  if (MODE == 2) {
    float* o = (float*)outp;
    #pragma unroll
    for (int t = 0; t < 4; ++t) {
      int col = n0b + t * 16 + nlane;
      if (col >= N) continue;
      float bc = bias[col];
      #pragma unroll
      for (int v = 0; v < 8; ++v) {
        int r = row0 + half * 8 + v;
        o[(size_t)r * N + col] = acc[t][v] + bc;
      }
    }
  } else {
    bf16_t* o = (bf16_t*)outp;
    #pragma unroll
    for (int t = 0; t < 4; ++t) {
      int col = n0b + t * 16 + nlane;
      if (col >= N) continue;
      float bc = bias[col];
      float t0 = 0.f, t1 = 0.f, t2 = 0.f, t3 = 0.f;
      if (MODE == 1) {
        t0 = tail[col * 4 + 0]; t1 = tail[col * 4 + 1];
        t2 = tail[col * 4 + 2]; t3 = tail[col * 4 + 3];
      }
      #pragma unroll
      for (int v = 0; v < 8; ++v) {
        int r = row0 + half * 8 + v;
        float x = acc[t][v] + bc;
        if (MODE == 1) x += s0[v] * t0 + s1[v] * t1 + s2[v] * t2 + pf * t3;
        x = x / (1.0f + __expf(-x));   // silu
        o[(size_t)r * N + col] = (bf16_t)x;
      }
    }
  }
}

// ---------------------------------------------------------------------------
// Finalize: one wave per token. Mode logits from c (dot with cw), softmax-4,
// stage bias, banded-masked 7x7 softmax. Writes K (49) and alpha (4).
// ---------------------------------------------------------------------------
__global__ __launch_bounds__(32)
void k_finalize(const float* __restrict__ cvec, const float* __restrict__ baseb,
                const float* __restrict__ cw, const float* __restrict__ constl,
                const float* __restrict__ ms_w, const int* __restrict__ stage_graph,
                float* __restrict__ outK, float* __restrict__ outAlpha) {
  int t = blockIdx.x, lane = threadIdx.x;
  const float* c = cvec + (size_t)t * CH;

  float p0 = 0.f, p1 = 0.f, p2 = 0.f, p3 = 0.f;
  for (int j = lane; j < CH; j += 32) {
    float cv = c[j];
    p0 += cv * cw[0 * CH + j];
    p1 += cv * cw[1 * CH + j];
    p2 += cv * cw[2 * CH + j];
    p3 += cv * cw[3 * CH + j];
  }
  #pragma unroll
  for (int off = 16; off > 0; off >>= 1) {
    p0 += __shfl_xor(p0, off, 32);
    p1 += __shfl_xor(p1, off, 32);
    p2 += __shfl_xor(p2, off, 32);
    p3 += __shfl_xor(p3, off, 32);
  }
  float e0 = p0 + constl[0], e1 = p1 + constl[1];
  float e2 = p2 + constl[2], e3 = p3 + constl[3];
  float mx = fmaxf(fmaxf(e0, e1), fmaxf(e2, e3));
  float x0 = __expf(e0 - mx), x1 = __expf(e1 - mx);
  float x2 = __expf(e2 - mx), x3 = __expf(e3 - mx);
  float inv = 1.0f / (x0 + x1 + x2 + x3);
  float a0 = x0 * inv, a1 = x1 * inv, a2 = x2 * inv, a3 = x3 * inv;

  if (lane == 0) {
    outAlpha[t * 4 + 0] = a0; outAlpha[t * 4 + 1] = a1;
    outAlpha[t * 4 + 2] = a2; outAlpha[t * 4 + 3] = a3;
  }

  if (lane < NSTAGES) {
    int i = lane;
    const float* brow = baseb + (size_t)t * (NSTAGES * NSTAGES) + i * NSTAGES;
    float raws[NSTAGES];
    int msk[NSTAGES];
    float rmax = -3.4e38f;
    #pragma unroll
    for (int j = 0; j < NSTAGES; ++j) {
      msk[j] = stage_graph[i * NSTAGES + j];
      if (msk[j]) {
        float sb = a0 * ms_w[j * NMODES + 0] + a1 * ms_w[j * NMODES + 1] +
                   a2 * ms_w[j * NMODES + 2] + a3 * ms_w[j * NMODES + 3];
        raws[j] = brow[j] + sb;
        rmax = fmaxf(rmax, raws[j]);
      }
    }
    float Z = 0.f;
    #pragma unroll
    for (int j = 0; j < NSTAGES; ++j) {
      if (msk[j]) { raws[j] = __expf(raws[j] - rmax); Z += raws[j]; }
    }
    float iZ = 1.0f / Z;
    float* orow = outK + (size_t)t * (NSTAGES * NSTAGES) + i * NSTAGES;
    #pragma unroll
    for (int j = 0; j < NSTAGES; ++j) orow[j] = msk[j] ? raws[j] * iZ : 0.0f;
  }
}

// ---------------------------------------------------------------------------
extern "C" void kernel_launch(void* const* d_in, const int* in_sizes, int n_in,
                              void* d_out, int out_size, void* d_ws, size_t ws_size,
                              hipStream_t stream) {
  (void)in_sizes; (void)n_in; (void)out_size; (void)ws_size;
  const float* h        = (const float*)d_in[0];
  const float* emb_x    = (const float*)d_in[1];
  const float* lam      = (const float*)d_in[2];
  const float* mu_prev  = (const float*)d_in[3];
  const float* base_w1  = (const float*)d_in[4];
  const float* base_b1  = (const float*)d_in[5];
  const float* base_w2  = (const float*)d_in[6];
  const float* base_b2  = (const float*)d_in[7];
  const float* ln_g     = (const float*)d_in[8];
  const float* ln_b     = (const float*)d_in[9];
  const float* mlp_w1   = (const float*)d_in[10];
  const float* mlp_b1   = (const float*)d_in[11];
  const float* mlp_w2   = (const float*)d_in[12];
  const float* mlp_b2   = (const float*)d_in[13];
  const float* pass_emb = (const float*)d_in[14];
  const float* ch_w     = (const float*)d_in[15];
  const float* ch_b     = (const float*)d_in[16];
  const float* ph_w     = (const float*)d_in[17];
  const float* ph_b     = (const float*)d_in[18];
  const float* bil_w    = (const float*)d_in[19];
  const float* mode_bias= (const float*)d_in[20];
  const float* ms_w     = (const float*)d_in[21];
  const int*   sgraph   = (const int*)d_in[22];
  const int*   pass_idx = (const int*)d_in[23];

  const int B = 8, T = 2048, D = 768, BT = B * T;
  const int H1 = 256, K2 = 49, K2PAD = 64;

  char* ws = (char*)d_ws;
  size_t off = 0;
  auto alloc = [&](size_t bytes) -> void* {
    void* p = ws + off;
    off = (off + bytes + 255) & ~(size_t)255;
    return p;
  };

  bf16_t* h_bf   = (bf16_t*)alloc((size_t)BT * D * 2);
  bf16_t* r_bf   = (bf16_t*)alloc((size_t)BT * 1536 * 2);
  bf16_t* act1   = (bf16_t*)alloc((size_t)BT * H1 * 2);
  bf16_t* act3   = (bf16_t*)alloc((size_t)BT * CH * 2);
  float*  baseb  = (float*)alloc((size_t)BT * K2 * 4);
  float*  cvec   = (float*)alloc((size_t)BT * CH * 4);
  float*  stats  = (float*)alloc((size_t)BT * 4 * 4);
  bf16_t* w1a_bf = (bf16_t*)alloc((size_t)H1 * D * 2);
  bf16_t* w2_bf  = (bf16_t*)alloc((size_t)K2PAD * H1 * 2);   // zero-padded rows
  bf16_t* mw1_bf = (bf16_t*)alloc((size_t)CH * 1536 * 2);
  bf16_t* mw2_bf = (bf16_t*)alloc((size_t)CH * CH * 2);
  float*  cw     = (float*)alloc((size_t)NMODES * CH * 4);
  float*  constl = (float*)alloc(64);
  float*  tail   = (float*)alloc((size_t)CH * 4 * 4);

  // weight conversions (pad base_w2 rows 49 -> 64 with zeros)
  {
    int tot = H1 * D;
    k_cvt_bf16<<<(tot + 255) / 256, 256, 0, stream>>>(base_w1, w1a_bf, H1, H1, D, D);
  }
  {
    int tot = K2PAD * H1;
    k_cvt_bf16<<<(tot + 255) / 256, 256, 0, stream>>>(base_w2, w2_bf, K2, K2PAD, H1, H1);
  }
  {
    int tot = CH * 1536;
    k_cvt_bf16<<<(tot + 255) / 256, 256, 0, stream>>>(mlp_w1, mw1_bf, CH, CH, 1540, 1536);
  }
  {
    int tot = CH * CH;
    k_cvt_bf16<<<(tot + 255) / 256, 256, 0, stream>>>(mlp_w2, mw2_bf, CH, CH, CH, CH);
  }

  k_prep_small<<<1, 128, 0, stream>>>(pass_emb, ch_w, ch_b, ph_w, ph_b, bil_w,
                                      mode_bias, mlp_w1, pass_idx, cw, constl, tail);

  k_prep_token<<<BT, 256, 0, stream>>>(h, emb_x, lam, mu_prev, ln_g, ln_b,
                                       h_bf, r_bf, stats);

  // GEMM1: (BT,768)x(256,768)^T -> silu -> act1 bf16
  k_gemm_wmma<0><<<dim3(BT / 16, H1 / 64), 32, 0, stream>>>(
      h_bf, w1a_bf, base_b1, (void*)act1, BT, H1, D, nullptr, nullptr, nullptr);

  // GEMM3: (BT,1536)x(128,1536)^T + stats-tail -> silu -> act3 bf16
  k_gemm_wmma<1><<<dim3(BT / 16, CH / 64), 32, 0, stream>>>(
      r_bf, mw1_bf, mlp_b1, (void*)act3, BT, CH, 1536, stats, tail, pass_idx);

  // GEMM2: (BT,256)x(64pad,256)^T -> f32 base (cols >= 49 not stored)
  k_gemm_wmma<2><<<dim3(BT / 16, K2PAD / 64), 32, 0, stream>>>(
      act1, w2_bf, base_b2, (void*)baseb, BT, K2, H1, nullptr, nullptr, nullptr);

  // GEMM4: (BT,128)x(128,128)^T -> f32 c
  k_gemm_wmma<2><<<dim3(BT / 16, CH / 64), 32, 0, stream>>>(
      act3, mw2_bf, mlp_b2, (void*)cvec, BT, CH, CH, nullptr, nullptr, nullptr);

  float* outK = (float*)d_out;
  float* outA = outK + (size_t)BT * (NSTAGES * NSTAGES);
  k_finalize<<<BT, 32, 0, stream>>>(cvec, baseb, cw, constl, ms_w, sgraph, outK, outA);
}
